// CausalSelfAttention_66589172957720
// MI455X (gfx1250) — matrix operand order repaired and checked by reading, not compiled, not run on hardware
//
#include <hip/hip_runtime.h>

#define Bq 4
#define Tq 2048
#define Cq 1024
#define Hq 16
#define Dq 64
#define Mq (Bq*Tq)      /* 8192 rows of x            */
#define Kq Cq           /* 1024 reduction dim        */
#define N_QKV (3*Cq)    /* 3072 qkv output cols      */

typedef __bf16 bf16;
typedef __attribute__((ext_vector_type(8)))  __bf16 v8bf;
typedef __attribute__((ext_vector_type(16))) __bf16 v16bf;
typedef __attribute__((ext_vector_type(8)))  float  v8f;

// ---------------------------------------------------------------------------
// WMMA fragment loaders (wave32, 16x16x32 bf16).
// A (16x32, MxK): lane holds row M=lane&15; elems 0..7 -> K = hi*8+0..7,
//                 elems 8..15 -> K = hi*8+16..23   (hi = lane>>4)
// B (32x16, KxN): lane holds col N=lane&15; elems 0..7 -> K = hi*8+0..7,
//                 elems 8..15 -> K = hi*8+16..23
// C/D (16x16 f32): 8 VGPRs; row = hi*8 + r, col = lane&15
// ---------------------------------------------------------------------------
__device__ __forceinline__ v16bf frag_a(const bf16* p, int hi) {
  v8bf lo = *reinterpret_cast<const v8bf*>(p + hi * 8);
  v8bf hh = *reinterpret_cast<const v8bf*>(p + hi * 8 + 16);
  v16bf f;
#pragma unroll
  for (int i = 0; i < 8; ++i) { f[i] = lo[i]; f[i + 8] = hh[i]; }
  return f;
}

__device__ __forceinline__ v16bf frag_b(const bf16* p, int hi) {
  v8bf lo = *reinterpret_cast<const v8bf*>(p + hi * 8);
  v8bf hh = *reinterpret_cast<const v8bf*>(p + hi * 8 + 16);
  v16bf f;
#pragma unroll
  for (int i = 0; i < 8; ++i) { f[i] = lo[i]; f[i + 8] = hh[i]; }
  return f;
}

__device__ __forceinline__ v8f wmma_bf16(v16bf a, v16bf b, v8f c) {
  return __builtin_amdgcn_wmma_f32_16x16x32_bf16(
      /*neg_a=*/false, a, /*neg_b=*/false, b,
      /*c_mod=*/(short)0, c, /*reuse_a=*/false, /*reuse_b=*/false);
}

// ---------------------------------------------------------------------------
// fp32 -> bf16 cast / transpose-cast helpers
// ---------------------------------------------------------------------------
__global__ void cast_bf16_kernel(const float* __restrict__ src,
                                 bf16* __restrict__ dst, int n) {
  int i = blockIdx.x * blockDim.x + threadIdx.x;
  if (i < n) dst[i] = (bf16)src[i];
}

// src is [Krows, Ncols] row-major; dst is [Ncols, Krows] (i.e. W^T) bf16
__global__ void transpose_cast_kernel(const float* __restrict__ src,
                                      bf16* __restrict__ dst,
                                      int Krows, int Ncols) {
  int i = blockIdx.x * blockDim.x + threadIdx.x;
  if (i >= Krows * Ncols) return;
  int kr = i / Ncols;
  int n  = i % Ncols;
  dst[n * Krows + kr] = (bf16)src[i];
}

// ---------------------------------------------------------------------------
// QKV GEMM: [M,K] bf16 x [N,K]^T bf16 -> scattered q [B,H,T,D] (pre-scaled by
// 1/sqrt(D)), k [B,H,T,D], v^T [B,H,D,T], all bf16.  64x64 tile per wave.
// __launch_bounds__(32,1): full VGPR file per wave -> no accumulator spills.
// ---------------------------------------------------------------------------
__global__ void __launch_bounds__(32, 1)
gemm_qkv_kernel(const bf16* __restrict__ A,
                const bf16* __restrict__ WT,
                const float* __restrict__ bias,
                bf16* __restrict__ qb,
                bf16* __restrict__ kb,
                bf16* __restrict__ vTb) {
  const int lane = threadIdx.x & 31;
  const int col  = lane & 15;
  const int hi   = lane >> 4;
  const int m0   = blockIdx.x * 64;
  const int n0   = blockIdx.y * 64;

  v8f acc[4][4] = {};
  for (int kk = 0; kk < Kq; kk += 32) {
    v16bf af[4], bfr[4];
#pragma unroll
    for (int mi = 0; mi < 4; ++mi)
      af[mi] = frag_a(A + (m0 + mi * 16 + col) * Kq + kk, hi);
#pragma unroll
    for (int ni = 0; ni < 4; ++ni)
      bfr[ni] = frag_b(WT + (n0 + ni * 16 + col) * Kq + kk, hi);
    if (kk + 32 < Kq) {   // pull next slice toward the WGP (global_prefetch_b8)
#pragma unroll
      for (int mi = 0; mi < 4; ++mi)
        __builtin_prefetch(A + (m0 + mi * 16 + col) * Kq + kk + 32, 0, 3);
#pragma unroll
      for (int ni = 0; ni < 4; ++ni)
        __builtin_prefetch(WT + (n0 + ni * 16 + col) * Kq + kk + 32, 0, 3);
    }
#pragma unroll
    for (int mi = 0; mi < 4; ++mi)
#pragma unroll
      for (int ni = 0; ni < 4; ++ni)
        acc[mi][ni] = wmma_bf16(af[mi], bfr[ni], acc[mi][ni]);
  }

#pragma unroll
  for (int mi = 0; mi < 4; ++mi)
#pragma unroll
    for (int ni = 0; ni < 4; ++ni)
#pragma unroll
      for (int e = 0; e < 8; ++e) {
        int gm = m0 + mi * 16 + hi * 8 + e;   // b*T + t
        int gn = n0 + ni * 16 + col;          // [0, 3C)
        float v = acc[mi][ni][e] + bias[gn];
        int bb  = gm >> 11;            // / T
        int t   = gm & (Tq - 1);
        int sec = gn >> 10;            // / C  -> 0=q,1=k,2=v
        int c   = gn & (Cq - 1);
        int h   = c >> 6;
        int d   = c & 63;
        if (sec == 0)
          qb[((bb * Hq + h) * Tq + t) * Dq + d] = (bf16)(v * 0.125f); // 1/sqrt(64)
        else if (sec == 1)
          kb[((bb * Hq + h) * Tq + t) * Dq + d] = (bf16)v;
        else
          vTb[((bb * Hq + h) * Dq + d) * Tq + t] = (bf16)v;
      }
}

// ---------------------------------------------------------------------------
// Flash attention: one wave = one 16-row query tile of one (b,h).
// Streams 32-key blocks; scores via 4 WMMAs, P·V via 4 WMMAs per block.
// Online softmax with fp32 running max/sum. Output y bf16 [B,T,C].
// ---------------------------------------------------------------------------
__global__ void __launch_bounds__(32, 1)
attn_kernel(const bf16* __restrict__ q,
            const bf16* __restrict__ kbuf,
            const bf16* __restrict__ vT,
            bf16* __restrict__ y) {
  __shared__ __align__(16) bf16 psh[16 * 32];

  const int lane = threadIdx.x & 31;
  const int col  = lane & 15;
  const int hi   = lane >> 4;
  const int q0   = blockIdx.x * 16;
  const int bh   = blockIdx.y;

  const bf16* qp = q    + (size_t)bh * Tq * Dq;
  const bf16* kp = kbuf + (size_t)bh * Tq * Dq;
  const bf16* vp = vT   + (size_t)bh * Dq * Tq;

  // Q fragments for d-chunks [0,32) and [32,64) (Q already scaled)
  const v16bf qf0 = frag_a(qp + (q0 + col) * Dq + 0, hi);
  const v16bf qf1 = frag_a(qp + (q0 + col) * Dq + 32, hi);

  v8f acc[4] = {};
  float m[8], l[8];
#pragma unroll
  for (int e = 0; e < 8; ++e) { m[e] = -1e30f; l[e] = 0.f; }

  for (int k0 = 0; k0 <= q0 + 15; k0 += 32) {
    // ---- scores: two 16x16 tiles over this 32-key block -------------------
    v8f s[2];
#pragma unroll
    for (int ti = 0; ti < 2; ++ti) {
      const bf16* kbase = kp + (k0 + ti * 16 + col) * Dq;
      v16bf b0 = frag_b(kbase + 0, hi);
      v16bf b1 = frag_b(kbase + 32, hi);
      v8f z = {};
      z = wmma_bf16(qf0, b0, z);
      s[ti] = wmma_bf16(qf1, b1, z);
    }
    // ---- causal mask (only on diagonal blocks) ----------------------------
    if (k0 + 31 > q0) {
#pragma unroll
      for (int ti = 0; ti < 2; ++ti)
#pragma unroll
        for (int e = 0; e < 8; ++e) {
          int kidx = k0 + ti * 16 + col;
          int row  = q0 + hi * 8 + e;
          if (kidx > row) s[ti][e] = -1e30f;
        }
    }
    // ---- online softmax ---------------------------------------------------
    float bm[8], sc[8], rs[8];
#pragma unroll
    for (int e = 0; e < 8; ++e) bm[e] = fmaxf(s[0][e], s[1][e]);
#pragma unroll
    for (int off = 8; off; off >>= 1)
#pragma unroll
      for (int e = 0; e < 8; ++e)
        bm[e] = fmaxf(bm[e], __shfl_xor(bm[e], off));
#pragma unroll
    for (int e = 0; e < 8; ++e) {
      float mn = fmaxf(m[e], bm[e]);
      sc[e] = __expf(m[e] - mn);
      m[e]  = mn;
      float p0 = __expf(s[0][e] - mn);
      float p1 = __expf(s[1][e] - mn);
      s[0][e] = p0; s[1][e] = p1;
      rs[e] = p0 + p1;
    }
#pragma unroll
    for (int off = 8; off; off >>= 1)
#pragma unroll
      for (int e = 0; e < 8; ++e)
        rs[e] += __shfl_xor(rs[e], off);
#pragma unroll
    for (int e = 0; e < 8; ++e) l[e] = l[e] * sc[e] + rs[e];
#pragma unroll
    for (int ni = 0; ni < 4; ++ni)
#pragma unroll
      for (int e = 0; e < 8; ++e) acc[ni][e] *= sc[e];

    // ---- relayout P: C-layout -> A-fragment via LDS -----------------------
    __syncthreads();
#pragma unroll
    for (int ti = 0; ti < 2; ++ti)
#pragma unroll
      for (int e = 0; e < 8; ++e)
        psh[(hi * 8 + e) * 32 + ti * 16 + col] = (bf16)s[ti][e];
    __syncthreads();
    v16bf pf = frag_a(&psh[col * 32], hi);

    // ---- y += P @ V -------------------------------------------------------
#pragma unroll
    for (int ni = 0; ni < 4; ++ni) {
      v16bf vf = frag_b(vp + (ni * 16 + col) * Tq + k0, hi);
      acc[ni] = wmma_bf16(pf, vf, acc[ni]);
    }
  }

  // ---- finalize: divide by row sums, store y [B,T,C] bf16 -----------------
  const int bb = bh >> 4;   // / H
  const int h  = bh & 15;
#pragma unroll
  for (int ni = 0; ni < 4; ++ni)
#pragma unroll
    for (int e = 0; e < 8; ++e) {
      int t = q0 + hi * 8 + e;
      float o = acc[ni][e] / l[e];
      y[((size_t)(bb * Tq + t)) * Cq + h * Dq + ni * 16 + col] = (bf16)o;
    }
}

// ---------------------------------------------------------------------------
// Output projection: y[M,C] bf16 x Wp^T[C,C] bf16 + bias -> out f32 [M,C]
// ---------------------------------------------------------------------------
__global__ void __launch_bounds__(32, 1)
gemm_proj_kernel(const bf16* __restrict__ A,
                 const bf16* __restrict__ WT,
                 const float* __restrict__ bias,
                 float* __restrict__ out) {
  const int lane = threadIdx.x & 31;
  const int col  = lane & 15;
  const int hi   = lane >> 4;
  const int m0   = blockIdx.x * 64;
  const int n0   = blockIdx.y * 64;

  v8f acc[4][4] = {};
  for (int kk = 0; kk < Kq; kk += 32) {
    v16bf af[4], bfr[4];
#pragma unroll
    for (int mi = 0; mi < 4; ++mi)
      af[mi] = frag_a(A + (m0 + mi * 16 + col) * Kq + kk, hi);
#pragma unroll
    for (int ni = 0; ni < 4; ++ni)
      bfr[ni] = frag_b(WT + (n0 + ni * 16 + col) * Kq + kk, hi);
    if (kk + 32 < Kq) {
#pragma unroll
      for (int mi = 0; mi < 4; ++mi)
        __builtin_prefetch(A + (m0 + mi * 16 + col) * Kq + kk + 32, 0, 3);
#pragma unroll
      for (int ni = 0; ni < 4; ++ni)
        __builtin_prefetch(WT + (n0 + ni * 16 + col) * Kq + kk + 32, 0, 3);
    }
#pragma unroll
    for (int mi = 0; mi < 4; ++mi)
#pragma unroll
      for (int ni = 0; ni < 4; ++ni)
        acc[mi][ni] = wmma_bf16(af[mi], bfr[ni], acc[mi][ni]);
  }

#pragma unroll
  for (int mi = 0; mi < 4; ++mi)
#pragma unroll
    for (int ni = 0; ni < 4; ++ni)
#pragma unroll
      for (int e = 0; e < 8; ++e) {
        int gm = m0 + mi * 16 + hi * 8 + e;
        int gn = n0 + ni * 16 + col;
        out[(size_t)gm * Cq + gn] = acc[mi][ni][e] + bias[gn];
      }
}

// ---------------------------------------------------------------------------
extern "C" void kernel_launch(void* const* d_in, const int* in_sizes, int n_in,
                              void* d_out, int out_size, void* d_ws, size_t ws_size,
                              hipStream_t stream) {
  const float* x  = (const float*)d_in[0];
  const float* Wa = (const float*)d_in[1];
  const float* ba = (const float*)d_in[2];
  const float* Wp = (const float*)d_in[3];
  const float* bp = (const float*)d_in[4];
  float* out = (float*)d_out;

  char* ws = (char*)d_ws;
  size_t off = 0;
  auto alloc = [&](size_t bytes) -> void* {
    void* p = ws + off;
    off += (bytes + 255) & ~(size_t)255;
    return p;
  };
  bf16* xb  = (bf16*)alloc(sizeof(bf16) * (size_t)Mq * Kq);
  bf16* WaT = (bf16*)alloc(sizeof(bf16) * (size_t)N_QKV * Kq);
  bf16* WpT = (bf16*)alloc(sizeof(bf16) * (size_t)Cq * Cq);
  bf16* qb  = (bf16*)alloc(sizeof(bf16) * (size_t)Bq * Hq * Tq * Dq);
  bf16* kb  = (bf16*)alloc(sizeof(bf16) * (size_t)Bq * Hq * Tq * Dq);
  bf16* vTb = (bf16*)alloc(sizeof(bf16) * (size_t)Bq * Hq * Dq * Tq);
  bf16* yb  = (bf16*)alloc(sizeof(bf16) * (size_t)Mq * Cq);

  int nX = Mq * Kq;
  cast_bf16_kernel<<<(nX + 255) / 256, 256, 0, stream>>>(x, xb, nX);
  transpose_cast_kernel<<<((Kq * N_QKV) + 255) / 256, 256, 0, stream>>>(Wa, WaT, Kq, N_QKV);
  transpose_cast_kernel<<<((Kq * Cq) + 255) / 256, 256, 0, stream>>>(Wp, WpT, Kq, Cq);

  gemm_qkv_kernel<<<dim3(Mq / 64, N_QKV / 64), 32, 0, stream>>>(xb, WaT, ba, qb, kb, vTb);
  attn_kernel<<<dim3(Tq / 16, Bq * Hq), 32, 0, stream>>>(qb, kb, vTb, yb);
  gemm_proj_kernel<<<dim3(Mq / 64, Cq / 64), 32, 0, stream>>>(yb, WpT, bp, out);
}
